// SAM3ViTLayer_54236847013894
// MI455X (gfx1250) — matrix operand order repaired and checked by reading, not compiled, not run on hardware
//
#include <hip/hip_runtime.h>

// ---------------- problem constants ----------------
#define SS   5184      // H*W tokens
#define WW   72
#define CC   768
#define FFD  3072
#define NHH  12
#define HDD  64

// ---------------- WMMA fragment types ----------------
typedef __attribute__((ext_vector_type(16))) __bf16 bf16x16;
typedef __attribute__((ext_vector_type(8)))  float  f32x8;

union ABfrag { bf16x16 v; uint4 q[2]; unsigned short h[16]; };

__device__ __forceinline__ unsigned short f2bf(float f) {
  unsigned int u = __float_as_uint(f);
  u += 0x7FFFu + ((u >> 16) & 1u);        // round-to-nearest-even
  return (unsigned short)(u >> 16);
}

// A-matrix 16x32 bf16 fragment (rows = lane&15, K contiguous in memory).
__device__ __forceinline__ bf16x16 load_afrag(const unsigned short* row, int kb, int hi) {
  ABfrag u;
  u.q[0] = *(const uint4*)(row + kb + hi * 16);
  u.q[1] = *(const uint4*)(row + kb + hi * 16 + 8);
  return u.v;
}
// B-matrix 32x16 bf16 fragment from [N][K] row-major (col = lane&15).
__device__ __forceinline__ bf16x16 load_bfrag(const unsigned short* row, int kb, int hi) {
  ABfrag u;
  u.q[0] = *(const uint4*)(row + kb + hi * 16);
  u.q[1] = *(const uint4*)(row + kb + hi * 16 + 8);
  return u.v;
}

__device__ __forceinline__ f32x8 wmma_bf16(bf16x16 a, bf16x16 b, f32x8 c) {
  return __builtin_amdgcn_wmma_f32_16x16x32_bf16(false, a, false, b, (short)0, c, false, false);
}

// D/C frag mapping: element r -> row M = r + (lane>>4)*8 , col N = lane&15

// ---------------- prep kernels ----------------
__global__ void __launch_bounds__(256)
transpose_bf16_kernel(const float* __restrict__ src, unsigned short* __restrict__ dst,
                      int K, int N) {
  __shared__ float t[16][17];
  int nb = blockIdx.x * 16, kb = blockIdx.y * 16;
  int tx = threadIdx.x, ty = threadIdx.y;
  t[ty][tx] = src[(size_t)(kb + ty) * N + nb + tx];
  __syncthreads();
  dst[(size_t)(nb + ty) * K + kb + tx] = f2bf(t[tx][ty]);
}

__global__ void __launch_bounds__(256)
rope_table_kernel(float* __restrict__ ct, float* __restrict__ st) {
  int idx = blockIdx.x * 256 + threadIdx.x;
  if (idx >= SS * 32) return;
  int s = idx >> 5, i = idx & 31;
  int y = s / WW, x = s % WW;
  float fr  = __powf(10000.0f, -(float)(i & 15) * (1.0f / 16.0f));
  float ang = (float)((i < 16) ? y : x) * fr;
  ct[idx] = cosf(ang);
  st[idx] = sinf(ang);
}

__global__ void __launch_bounds__(256)
ln_kernel(const float* __restrict__ x, const float* __restrict__ g,
          const float* __restrict__ b, unsigned short* __restrict__ out) {
  __shared__ float red[256];
  int row = blockIdx.x, tid = threadIdx.x;
  const float* xr = x + (size_t)row * CC;
  float v0 = xr[tid], v1 = xr[tid + 256], v2 = xr[tid + 512];
  red[tid] = v0 + v1 + v2;
  __syncthreads();
  for (int o = 128; o > 0; o >>= 1) { if (tid < o) red[tid] += red[tid + o]; __syncthreads(); }
  float mean = red[0] * (1.0f / CC);
  __syncthreads();
  float d0 = v0 - mean, d1 = v1 - mean, d2 = v2 - mean;
  red[tid] = d0 * d0 + d1 * d1 + d2 * d2;
  __syncthreads();
  for (int o = 128; o > 0; o >>= 1) { if (tid < o) red[tid] += red[tid + o]; __syncthreads(); }
  float rstd = rsqrtf(red[0] * (1.0f / CC) + 1e-6f);
  unsigned short* orow = out + (size_t)row * CC;
  orow[tid]       = f2bf(d0 * rstd * g[tid]       + b[tid]);
  orow[tid + 256] = f2bf(d1 * rstd * g[tid + 256] + b[tid + 256]);
  orow[tid + 512] = f2bf(d2 * rstd * g[tid + 512] + b[tid + 512]);
}

// ---------------- WMMA GEMM core: one wave -> 32(M) x 64(N), double-buffered ----------------
// acc[t*4+j] covers rows m0+t*16.., cols n0+j*16..
template<int KDIM>
__device__ __forceinline__ void gemm_core(const unsigned short* __restrict__ A,
                                          const unsigned short* __restrict__ WT,
                                          int m0, int n0, int lane, f32x8 acc[8]) {
  const int q = lane & 15, hi = lane >> 4;
  const unsigned short* arow0 = A + (size_t)(m0 + q) * KDIM;
  const unsigned short* arow1 = arow0 + (size_t)16 * KDIM;
  const unsigned short* br0 = WT + (size_t)(n0 + q) * KDIM;
  const unsigned short* br1 = br0 + (size_t)16 * KDIM;
  const unsigned short* br2 = br0 + (size_t)32 * KDIM;
  const unsigned short* br3 = br0 + (size_t)48 * KDIM;

  bf16x16 a0 = load_afrag(arow0, 0, hi);
  bf16x16 a1 = load_afrag(arow1, 0, hi);
  bf16x16 b0 = load_bfrag(br0, 0, hi);
  bf16x16 b1 = load_bfrag(br1, 0, hi);
  bf16x16 b2 = load_bfrag(br2, 0, hi);
  bf16x16 b3 = load_bfrag(br3, 0, hi);

#pragma unroll 2
  for (int kb = 0; kb < KDIM; kb += 32) {
    const int kn = (kb + 32 < KDIM) ? kb + 32 : 0;   // prefetch next (wraps harmlessly)
    bf16x16 a0n = load_afrag(arow0, kn, hi);
    bf16x16 a1n = load_afrag(arow1, kn, hi);
    bf16x16 b0n = load_bfrag(br0, kn, hi);
    bf16x16 b1n = load_bfrag(br1, kn, hi);
    bf16x16 b2n = load_bfrag(br2, kn, hi);
    bf16x16 b3n = load_bfrag(br3, kn, hi);

    acc[0] = wmma_bf16(a0, b0, acc[0]);
    acc[1] = wmma_bf16(a0, b1, acc[1]);
    acc[2] = wmma_bf16(a0, b2, acc[2]);
    acc[3] = wmma_bf16(a0, b3, acc[3]);
    acc[4] = wmma_bf16(a1, b0, acc[4]);
    acc[5] = wmma_bf16(a1, b1, acc[5]);
    acc[6] = wmma_bf16(a1, b2, acc[6]);
    acc[7] = wmma_bf16(a1, b3, acc[7]);

    a0 = a0n; a1 = a1n; b0 = b0n; b1 = b1n; b2 = b2n; b3 = b3n;
  }
}

__device__ __forceinline__ void zero_acc(f32x8 acc[8]) {
#pragma unroll
  for (int j = 0; j < 8; ++j)
#pragma unroll
    for (int r = 0; r < 8; ++r) acc[j][r] = 0.0f;
}

// Q/K projection: bias + RoPE, store [NH][S][64] bf16
__global__ void __launch_bounds__(32)
gemm_qk_rope_kernel(const unsigned short* __restrict__ A, const unsigned short* __restrict__ WT,
                    const float* __restrict__ bias, const float* __restrict__ ct,
                    const float* __restrict__ st, unsigned short* __restrict__ out) {
  int lane = threadIdx.x;
  int m0 = blockIdx.x * 32, n0 = blockIdx.y * 64;
  f32x8 acc[8];
  zero_acc(acc);
  gemm_core<CC>(A, WT, m0, n0, lane, acc);
  int q = lane & 15, hi = lane >> 4;
#pragma unroll
  for (int t = 0; t < 2; ++t)
#pragma unroll
    for (int j = 0; j < 4; ++j) {
      int cout = n0 + j * 16 + q;
      float bv = bias[cout];
      int hh = cout >> 6, c = cout & 63, i = c >> 1;
#pragma unroll
      for (int r = 0; r < 8; ++r) {
        int s = m0 + t * 16 + r + hi * 8;
        float val = acc[t * 4 + j][r] + bv;
        float partner = __shfl_xor(val, 1, 32);      // paired rope channel
        float rot = (c & 1) ? partner : -partner;
        float o = val * ct[s * 32 + i] + rot * st[s * 32 + i];
        out[(size_t)(hh * SS + s) * HDD + c] = f2bf(o);
      }
    }
}

// V projection: bias, store TRANSPOSED [NH][64][S] bf16
__global__ void __launch_bounds__(32)
gemm_v_kernel(const unsigned short* __restrict__ A, const unsigned short* __restrict__ WT,
              const float* __restrict__ bias, unsigned short* __restrict__ vt) {
  int lane = threadIdx.x;
  int m0 = blockIdx.x * 32, n0 = blockIdx.y * 64;
  f32x8 acc[8];
  zero_acc(acc);
  gemm_core<CC>(A, WT, m0, n0, lane, acc);
  int q = lane & 15, hi = lane >> 4;
#pragma unroll
  for (int t = 0; t < 2; ++t)
#pragma unroll
    for (int j = 0; j < 4; ++j) {
      int cout = n0 + j * 16 + q;
      float bv = bias[cout];
      int hh = cout >> 6, c = cout & 63;
#pragma unroll
      for (int r = 0; r < 8; ++r) {
        int s = m0 + t * 16 + r + hi * 8;
        vt[(size_t)(hh * HDD + c) * SS + s] = f2bf(acc[t * 4 + j][r] + bv);
      }
    }
}

// ---------------- flash attention: one wave per (head, 16-query block) ----------------
// S^T = K·Q^T (softmax reduction in-lane), O^T += V^T·P^T ; K/V tiles double-buffered.
__global__ void __launch_bounds__(32)
attn_kernel(const unsigned short* __restrict__ Qr, const unsigned short* __restrict__ Kb,
            const unsigned short* __restrict__ Vt, unsigned short* __restrict__ Ao) {
  const int lane = threadIdx.x;
  const int q = lane & 15, hi = lane >> 4;
  const int h = blockIdx.y;
  const int qb = blockIdx.x * 16;

  const unsigned short* qrow = Qr + (size_t)(h * SS + qb + q) * HDD;
  bf16x16 bq0 = load_bfrag(qrow, 0, hi);    // d = 0..31
  bf16x16 bq1 = load_bfrag(qrow, 32, hi);   // d = 32..63

  f32x8 acc[4];
#pragma unroll
  for (int m = 0; m < 4; ++m)
#pragma unroll
    for (int r = 0; r < 8; ++r) acc[m][r] = 0.0f;
  float mrun = -3.0e38f, lrun = 0.0f;

  const unsigned short* kbase = Kb + (size_t)h * SS * HDD;
  const unsigned short* vbase = Vt + (size_t)h * HDD * SS;

  auto loadK = [&](int kb, bf16x16 out[4]) {
    const unsigned short* krow0 = kbase + (size_t)(kb + q) * HDD;
    const unsigned short* krow1 = kbase + (size_t)(kb + 16 + q) * HDD;
    out[0] = load_afrag(krow0, 0, hi);
    out[1] = load_afrag(krow0, 32, hi);
    out[2] = load_afrag(krow1, 0, hi);
    out[3] = load_afrag(krow1, 32, hi);
  };
  auto loadV = [&](int kb, bf16x16 out[4]) {
#pragma unroll
    for (int m = 0; m < 4; ++m)
      out[m] = load_afrag(vbase + (size_t)(m * 16 + q) * SS, kb, hi);
  };

  bf16x16 ka[4], va[4], kan[4], van[4];
  loadK(0, ka);
  loadV(0, va);

  for (int kb = 0; kb < SS; kb += 32) {
    const int kn = (kb + 32 < SS) ? kb + 32 : 0;
    loadK(kn, kan);                      // prefetch next tile while computing this one
    loadV(kn, van);

    f32x8 s0, s1;
#pragma unroll
    for (int r = 0; r < 8; ++r) { s0[r] = 0.0f; s1[r] = 0.0f; }
    s0 = wmma_bf16(ka[0], bq0, s0);
    s0 = wmma_bf16(ka[1], bq1, s0);
    s1 = wmma_bf16(ka[2], bq0, s1);
    s1 = wmma_bf16(ka[3], bq1, s1);

    float sc0[8], sc1[8];
    float tmax = -3.0e38f;
#pragma unroll
    for (int r = 0; r < 8; ++r) {
      sc0[r] = s0[r] * 0.125f;           // HD^-0.5
      sc1[r] = s1[r] * 0.125f;
      tmax = fmaxf(tmax, fmaxf(sc0[r], sc1[r]));
    }
    tmax = fmaxf(tmax, __shfl_xor(tmax, 16, 32));
    float mnew  = fmaxf(mrun, tmax);
    float alpha = __expf(mrun - mnew);
    float p0[8], p1[8], rsum = 0.0f;
#pragma unroll
    for (int r = 0; r < 8; ++r) {
      p0[r] = __expf(sc0[r] - mnew);
      p1[r] = __expf(sc1[r] - mnew);
      rsum += p0[r] + p1[r];
    }
    rsum += __shfl_xor(rsum, 16, 32);
    lrun = lrun * alpha + rsum;
    mrun = mnew;
#pragma unroll
    for (int m = 0; m < 4; ++m)
#pragma unroll
      for (int r = 0; r < 8; ++r) acc[m][r] *= alpha;

    // repack P^T (32 keys x 16 queries) into a B fragment via one xor-16 exchange
    ABfrag pf;
#pragma unroll
    for (int r = 0; r < 8; ++r) {
      float send = hi ? p0[r] : p1[r];
      float rec  = __shfl_xor(send, 16, 32);
      pf.h[r]     = f2bf(hi ? rec   : p0[r]);   // keys base+16*hi+r (col q)
      pf.h[r + 8] = f2bf(hi ? p1[r] : rec);     // keys base+16*hi+8+r
    }
#pragma unroll
    for (int m = 0; m < 4; ++m)
      acc[m] = wmma_bf16(va[m], pf.v, acc[m]);

#pragma unroll
    for (int i = 0; i < 4; ++i) { ka[i] = kan[i]; va[i] = van[i]; }
  }
  float inv = 1.0f / lrun;
#pragma unroll
  for (int m = 0; m < 4; ++m)
#pragma unroll
    for (int r = 0; r < 8; ++r) {
      int d = m * 16 + r + hi * 8;
      Ao[(size_t)(qb + q) * CC + h * HDD + d] = f2bf(acc[m][r] * inv);
    }
}

// output projection: residual + layer-scale -> fp32 res1
__global__ void __launch_bounds__(32)
gemm_o_kernel(const unsigned short* __restrict__ A, const unsigned short* __restrict__ WT,
              const float* __restrict__ bias, const float* __restrict__ ls,
              const float* __restrict__ resid, float* __restrict__ out) {
  int lane = threadIdx.x;
  int m0 = blockIdx.x * 32, n0 = blockIdx.y * 64;
  f32x8 acc[8];
  zero_acc(acc);
  gemm_core<CC>(A, WT, m0, n0, lane, acc);
  int q = lane & 15, hi = lane >> 4;
#pragma unroll
  for (int t = 0; t < 2; ++t)
#pragma unroll
    for (int j = 0; j < 4; ++j) {
      int cout = n0 + j * 16 + q;
      float bv = bias[cout], lsv = ls[cout];
#pragma unroll
      for (int r = 0; r < 8; ++r) {
        int s = m0 + t * 16 + r + hi * 8;
        size_t idx = (size_t)s * CC + cout;
        out[idx] = resid[idx] + (acc[t * 4 + j][r] + bv) * lsv;
      }
    }
}

// FC1 + exact GELU -> bf16 [S][3072]
__global__ void __launch_bounds__(32)
gemm_fc1_kernel(const unsigned short* __restrict__ A, const unsigned short* __restrict__ WT,
                const float* __restrict__ bias, unsigned short* __restrict__ out) {
  int lane = threadIdx.x;
  int m0 = blockIdx.x * 32, n0 = blockIdx.y * 64;
  f32x8 acc[8];
  zero_acc(acc);
  gemm_core<CC>(A, WT, m0, n0, lane, acc);
  int q = lane & 15, hi = lane >> 4;
#pragma unroll
  for (int t = 0; t < 2; ++t)
#pragma unroll
    for (int j = 0; j < 4; ++j) {
      int cout = n0 + j * 16 + q;
      float bv = bias[cout];
#pragma unroll
      for (int r = 0; r < 8; ++r) {
        int s = m0 + t * 16 + r + hi * 8;
        float v = acc[t * 4 + j][r] + bv;
        float ge = 0.5f * v * (1.0f + erff(v * 0.70710678f));
        out[(size_t)s * FFD + cout] = f2bf(ge);
      }
    }
}

// FC2 + residual + layer-scale -> fp32 output
__global__ void __launch_bounds__(32)
gemm_fc2_kernel(const unsigned short* __restrict__ A, const unsigned short* __restrict__ WT,
                const float* __restrict__ bias, const float* __restrict__ ls,
                const float* __restrict__ resid, float* __restrict__ out) {
  int lane = threadIdx.x;
  int m0 = blockIdx.x * 32, n0 = blockIdx.y * 64;
  f32x8 acc[8];
  zero_acc(acc);
  gemm_core<FFD>(A, WT, m0, n0, lane, acc);
  int q = lane & 15, hi = lane >> 4;
#pragma unroll
  for (int t = 0; t < 2; ++t)
#pragma unroll
    for (int j = 0; j < 4; ++j) {
      int cout = n0 + j * 16 + q;
      float bv = bias[cout], lsv = ls[cout];
#pragma unroll
      for (int r = 0; r < 8; ++r) {
        int s = m0 + t * 16 + r + hi * 8;
        size_t idx = (size_t)s * CC + cout;
        out[idx] = resid[idx] + (acc[t * 4 + j][r] + bv) * lsv;
      }
    }
}

// ---------------- host-side launcher ----------------
extern "C" void kernel_launch(void* const* d_in, const int* in_sizes, int n_in,
                              void* d_out, int out_size, void* d_ws, size_t ws_size,
                              hipStream_t stream) {
  (void)in_sizes; (void)n_in; (void)out_size; (void)ws_size;
  const float* hs    = (const float*)d_in[0];
  const float* wq    = (const float*)d_in[1];
  const float* bq_   = (const float*)d_in[2];
  const float* wk    = (const float*)d_in[3];
  const float* bk_   = (const float*)d_in[4];
  const float* wv    = (const float*)d_in[5];
  const float* bv_   = (const float*)d_in[6];
  const float* wo    = (const float*)d_in[7];
  const float* bo_   = (const float*)d_in[8];
  const float* g1    = (const float*)d_in[9];
  const float* b1    = (const float*)d_in[10];
  const float* g2    = (const float*)d_in[11];
  const float* b2    = (const float*)d_in[12];
  const float* wfc1  = (const float*)d_in[13];
  const float* bfc1_ = (const float*)d_in[14];
  const float* wfc2  = (const float*)d_in[15];
  const float* bfc2_ = (const float*)d_in[16];
  const float* ls1   = (const float*)d_in[17];
  const float* ls2   = (const float*)d_in[18];
  float* out = (float*)d_out;

  char* p = (char*)d_ws;
  auto alloc = [&](size_t bytes) { char* r = p; p += (bytes + 255) & ~(size_t)255; return r; };
  unsigned short* WTq   = (unsigned short*)alloc((size_t)CC * CC * 2);
  unsigned short* WTk   = (unsigned short*)alloc((size_t)CC * CC * 2);
  unsigned short* WTv   = (unsigned short*)alloc((size_t)CC * CC * 2);
  unsigned short* WTo   = (unsigned short*)alloc((size_t)CC * CC * 2);
  unsigned short* WTfc1 = (unsigned short*)alloc((size_t)CC * FFD * 2);
  unsigned short* WTfc2 = (unsigned short*)alloc((size_t)CC * FFD * 2);
  float*          ct    = (float*)alloc((size_t)SS * 32 * 4);
  float*          st    = (float*)alloc((size_t)SS * 32 * 4);
  unsigned short* xn1   = (unsigned short*)alloc((size_t)SS * CC * 2);
  unsigned short* Qr    = (unsigned short*)alloc((size_t)NHH * SS * HDD * 2);
  unsigned short* Kbuf  = (unsigned short*)alloc((size_t)NHH * SS * HDD * 2);
  unsigned short* Vt    = (unsigned short*)alloc((size_t)NHH * SS * HDD * 2);
  unsigned short* Ao    = (unsigned short*)alloc((size_t)SS * CC * 2);
  float*          res1  = (float*)alloc((size_t)SS * CC * 4);
  unsigned short* xn2   = (unsigned short*)alloc((size_t)SS * CC * 2);
  unsigned short* h1    = (unsigned short*)alloc((size_t)SS * FFD * 2);

  dim3 tb(16, 16);
  transpose_bf16_kernel<<<dim3(CC / 16, CC / 16), tb, 0, stream>>>(wq, WTq, CC, CC);
  transpose_bf16_kernel<<<dim3(CC / 16, CC / 16), tb, 0, stream>>>(wk, WTk, CC, CC);
  transpose_bf16_kernel<<<dim3(CC / 16, CC / 16), tb, 0, stream>>>(wv, WTv, CC, CC);
  transpose_bf16_kernel<<<dim3(CC / 16, CC / 16), tb, 0, stream>>>(wo, WTo, CC, CC);
  transpose_bf16_kernel<<<dim3(FFD / 16, CC / 16), tb, 0, stream>>>(wfc1, WTfc1, CC, FFD);
  transpose_bf16_kernel<<<dim3(CC / 16, FFD / 16), tb, 0, stream>>>(wfc2, WTfc2, FFD, CC);
  rope_table_kernel<<<(SS * 32 + 255) / 256, 256, 0, stream>>>(ct, st);

  ln_kernel<<<SS, 256, 0, stream>>>(hs, g1, b1, xn1);

  dim3 gq(SS / 32, CC / 64);                    // 162 x 12, one wave each, 32x64 tiles
  gemm_qk_rope_kernel<<<gq, 32, 0, stream>>>(xn1, WTq, bq_, ct, st, Qr);
  gemm_qk_rope_kernel<<<gq, 32, 0, stream>>>(xn1, WTk, bk_, ct, st, Kbuf);
  gemm_v_kernel<<<gq, 32, 0, stream>>>(xn1, WTv, bv_, Vt);

  attn_kernel<<<dim3(SS / 16, NHH), 32, 0, stream>>>(Qr, Kbuf, Vt, Ao);

  gemm_o_kernel<<<gq, 32, 0, stream>>>(Ao, WTo, bo_, ls1, hs, res1);
  ln_kernel<<<SS, 256, 0, stream>>>(res1, g2, b2, xn2);
  gemm_fc1_kernel<<<dim3(SS / 32, FFD / 64), 32, 0, stream>>>(xn2, WTfc1, bfc1_, h1);
  gemm_fc2_kernel<<<gq, 32, 0, stream>>>(h1, WTfc2, bfc2_, ls2, res1, out);
}
